// Model_83141976916087
// MI455X (gfx1250) — compile-verified, matrix operand order checked by reading, not run on hardware
//
#include <hip/hip_runtime.h>
#include <hip/hip_bf16.h>

// ---------------------------------------------------------------------------
// Types for WMMA bf16 (wave32, gfx1250)
// ---------------------------------------------------------------------------
typedef __attribute__((ext_vector_type(16))) __bf16 v16bf;
typedef __attribute__((ext_vector_type(8)))  float  v8f;

union Frag16 {               // one 16x32 (A) or 32x16 (B) bf16 fragment: 8 VGPRs
    v16bf        v;
    unsigned int u[8];
};

#define N_EMBD 768
#define N_HEAD 8
#define HS     96
#define BB     4
#define TT     2048
#define MTOT   (BB * TT)      // 8192

__device__ __forceinline__ unsigned short f2bf(float f) {
    unsigned int u = __float_as_uint(f);
    unsigned int r = u + 0x7FFFu + ((u >> 16) & 1u);   // round-to-nearest-even
    return (unsigned short)(r >> 16);
}

// u32 index (within a row of 32 bf16 = 16 dwords) of fragment register i
// A-frag lane L: row = L%16; reg i holds bf16 pair at k = (i<4?2i:16+2(i-4)) + 8*(L/16)
__device__ __forceinline__ int kidx(int i, int half) {
    return (i < 4 ? i : i + 4) + 4 * half;
}

// Async global->LDS copy, 16 bytes per lane (gfx1250 GLOBAL_LOAD_ASYNC_TO_LDS_B128).
// lds_off = per-lane LDS byte address (low 32 bits of generic pointer to __shared__).
__device__ __forceinline__ void async_copy16(unsigned lds_off, const void* gaddr) {
    asm volatile("global_load_async_to_lds_b128 %0, %1, off"
                 :: "v"(lds_off), "v"(gaddr) : "memory");
}
__device__ __forceinline__ void async_wait0() {
    asm volatile("s_wait_asynccnt 0" ::: "memory");
}

// ---------------------------------------------------------------------------
// fp32 -> bf16 bulk convert
// ---------------------------------------------------------------------------
__global__ void cvt_bf16_kernel(const float* __restrict__ src,
                                unsigned short* __restrict__ dst, int n) {
    int i = blockIdx.x * blockDim.x + threadIdx.x;
    int stride = gridDim.x * blockDim.x;
    for (; i < n; i += stride) dst[i] = f2bf(src[i]);
}

// ---------------------------------------------------------------------------
// fp32 W[K][N] -> bf16 Wt[N][K]  (tiled 32x32 transpose through LDS)
// grid (N/32, K/32), block (32,8)
// ---------------------------------------------------------------------------
__global__ __launch_bounds__(256)
void cvt_transpose_kernel(const float* __restrict__ W,
                          unsigned short* __restrict__ Wt, int K, int N) {
    __shared__ float tile[32][33];
    const int n0 = blockIdx.x * 32;
    const int k0 = blockIdx.y * 32;
    const int tx = threadIdx.x;          // 0..31
    const int ty = threadIdx.y;          // 0..7
#pragma unroll
    for (int j = 0; j < 4; ++j) {
        int k = ty + 8 * j;
        tile[k][tx] = W[(size_t)(k0 + k) * N + n0 + tx];
    }
    __syncthreads();
#pragma unroll
    for (int j = 0; j < 4; ++j) {
        int n = ty + 8 * j;
        Wt[(size_t)(n0 + n) * K + k0 + tx] = f2bf(tile[tx][n]);
    }
}

// ---------------------------------------------------------------------------
// Tiled bf16 GEMM:  D[M,N] = A[M,K] * Wt[N,K]^T + bias[N]
//   A row-major [M][K] bf16, Wt row-major [N][K] bf16 (pre-transposed weight)
//   block = 256 threads (8 waves), block tile 128x128, K-step 32
//   Staging: GLOBAL_LOAD_ASYNC_TO_LDS_B128 (no VGPR round-trip), ASYNCcnt
//   mode 0: scatter q/k [B,H,T,hs] and v [B,H,hs,T] bf16   (qkv projection)
//   mode 1: write fp32 to out [M,N]                        (output projection)
// ---------------------------------------------------------------------------
__global__ __launch_bounds__(256)
void gemm_bf16_kernel(const unsigned short* __restrict__ A,
                      const unsigned short* __restrict__ Wt,
                      const float* __restrict__ bias,
                      int N, int Kdim, int mode,
                      unsigned short* __restrict__ Qp,
                      unsigned short* __restrict__ Kp,
                      unsigned short* __restrict__ Vp,
                      float* __restrict__ outf) {
    alignas(16) __shared__ unsigned short Ash[128 * 32];   // [row][k]
    alignas(16) __shared__ unsigned short Bsh[128 * 32];   // [n][k]

    const int tid  = threadIdx.x;
    const int w    = tid >> 5;
    const int lane = tid & 31;
    const int lm   = lane & 15;
    const int half = lane >> 4;
    const int mb   = blockIdx.x * 128;
    const int nb   = blockIdx.y * 128;

    const unsigned aoff = (unsigned)(size_t)(void*)Ash;
    const unsigned boff = (unsigned)(size_t)(void*)Bsh;

    v8f acc[8];
#pragma unroll
    for (int nt = 0; nt < 8; ++nt)
#pragma unroll
        for (int r = 0; r < 8; ++r) acc[nt][r] = 0.0f;

    const unsigned int* Ash32 = (const unsigned int*)Ash;
    const unsigned int* Bsh32 = (const unsigned int*)Bsh;

    for (int k0 = 0; k0 < Kdim; k0 += 32) {
        // ---- stage A and B tiles (128x32 bf16 each) via async b128 copies ----
        // 512 16-byte chunks per tile; 2 per thread per tile
#pragma unroll
        for (int it = 0; it < 2; ++it) {
            int q   = tid + 256 * it;            // chunk id 0..511
            int row = q >> 2;                    // 4 chunks per 32-el row
            int c4  = q & 3;
            async_copy16(aoff + q * 16, A  + (size_t)(mb + row) * Kdim + k0 + c4 * 8);
            async_copy16(boff + q * 16, Wt + (size_t)(nb + row) * Kdim + k0 + c4 * 8);
        }
        // ---- prefetch next K-tile (global_prefetch_b8) ----
        if (k0 + 32 < Kdim) {
            int row = tid & 127;
            __builtin_prefetch(A  + (size_t)(mb + row) * Kdim + k0 + 32, 0, 0);
            __builtin_prefetch(Wt + (size_t)(nb + row) * Kdim + k0 + 32, 0, 0);
        }
        async_wait0();
        __syncthreads();

        // ---- A fragment for this wave's 16 rows ----
        Frag16 af;
#pragma unroll
        for (int i = 0; i < 8; ++i)
            af.u[i] = Ash32[(16 * w + lm) * 16 + kidx(i, half)];

        // ---- 8 WMMAs across the 128-wide N strip ----
#pragma unroll
        for (int nt = 0; nt < 8; ++nt) {
            Frag16 bf;
#pragma unroll
            for (int i = 0; i < 8; ++i)
                bf.u[i] = Bsh32[(nt * 16 + lm) * 16 + kidx(i, half)];
            acc[nt] = __builtin_amdgcn_wmma_f32_16x16x32_bf16(
                false, af.v, false, bf.v, (short)0, acc[nt], false, false);
        }
        __syncthreads();
    }

    // ---- epilogue ----
#pragma unroll
    for (int nt = 0; nt < 8; ++nt) {
#pragma unroll
        for (int r = 0; r < 8; ++r) {
            int row = mb + 16 * w + r + 8 * half;
            int col = nb + nt * 16 + lm;
            float v = acc[nt][r] + bias[col];
            if (mode == 0) {
                int which = col / N_EMBD;
                int c     = col - which * N_EMBD;
                int h     = c / HS;
                int d     = c - h * HS;
                int bi    = row >> 11;           // /2048
                int t     = row & 2047;
                unsigned short bv = f2bf(v);
                if (which == 0)
                    Qp[((size_t)(bi * N_HEAD + h) * TT + t) * HS + d] = bv;
                else if (which == 1)
                    Kp[((size_t)(bi * N_HEAD + h) * TT + t) * HS + d] = bv;
                else  // V stored transposed: [b,h,d,t] for coalesced attn staging
                    Vp[((size_t)(bi * N_HEAD + h) * HS + d) * TT + t] = bv;
            } else {
                outf[(size_t)row * N_EMBD + col] = v;
            }
        }
    }
}

// ---------------------------------------------------------------------------
// Flash attention (causal), bf16 QKV -> bf16 Y[M, C]
//   Q,K: [B,H,T,hs]   V: [B,H,hs,T] (transposed)
//   grid (B*H, T/128); 8 waves; each wave = 16 query rows; key blocks of 32
// ---------------------------------------------------------------------------
__global__ __launch_bounds__(256)
void attn_kernel(const unsigned short* __restrict__ Qb,
                 const unsigned short* __restrict__ Kb,
                 const unsigned short* __restrict__ Vt,
                 unsigned short* __restrict__ Y) {
    alignas(16) __shared__ unsigned short Ksh[32 * HS];        // [key][d]
    alignas(16) __shared__ unsigned short Vtsh[HS * 32];       // [d][key]
    alignas(16) __shared__ unsigned short Psh[8 * 16 * 32];    // per-wave P (bf16)

    const int tid  = threadIdx.x;
    const int w    = tid >> 5;
    const int lane = tid & 31;
    const int lm   = lane & 15;
    const int half = lane >> 4;
    const int bh   = blockIdx.x;
    const int b    = bh >> 3;
    const int h    = bh & 7;
    const int q0b  = blockIdx.y * 128;
    const int q0   = q0b + 16 * w;
    const float sc = 0.1020620726159658f;          // 1/sqrt(96)

    const unsigned koff = (unsigned)(size_t)(void*)Ksh;
    const unsigned voff = (unsigned)(size_t)(void*)Vtsh;

    // Q fragments (3 K-steps of 32 over hs=96), loaded once
    Frag16 qf[3];
    {
        const unsigned int* qrow =
            (const unsigned int*)(Qb + ((size_t)bh * TT + q0 + lm) * HS);
#pragma unroll
        for (int j = 0; j < 3; ++j)
#pragma unroll
            for (int i = 0; i < 8; ++i)
                qf[j].u[i] = qrow[kidx(i, half) + 16 * j];
    }

    v8f acc[6];
    float m_s[8], l_s[8];
#pragma unroll
    for (int nt = 0; nt < 6; ++nt)
#pragma unroll
        for (int r = 0; r < 8; ++r) acc[nt][r] = 0.0f;
#pragma unroll
    for (int r = 0; r < 8; ++r) { m_s[r] = -__builtin_inff(); l_s[r] = 0.0f; }

    const unsigned int* Ksh32  = (const unsigned int*)Ksh;
    const unsigned int* Vtsh32 = (const unsigned int*)Vtsh;
    unsigned short* Pw = Psh + w * (16 * 32);

    const int nkb = q0b / 32 + 4;                  // causal bound for block
    for (int kb = 0; kb < nkb; ++kb) {
        // ---- stage K (32x96) and Vt (96x32) via async b128 copies ----
        // K: 384 chunks (12 per 96-el row); Vt: 384 chunks (4 per 32-el row)
#pragma unroll
        for (int it = 0; it < 3; ++it) {
            int e = tid + 256 * it;                // 0..767
            if (e < 384) {
                int row = e / 12;
                int c   = e - row * 12;
                async_copy16(koff + e * 16,
                             Kb + ((size_t)bh * TT + kb * 32 + row) * HS + c * 8);
            } else {
                int q = e - 384;
                int d = q >> 2;
                int c = q & 3;
                async_copy16(voff + q * 16,
                             Vt + ((size_t)bh * HS + d) * TT + kb * 32 + c * 8);
            }
        }
        async_wait0();
        __syncthreads();

        if (kb * 32 <= q0 + 15) {
            // ---- S = Q K^T for 32 keys (two 16x16 tiles) ----
            float st[2][8];
#pragma unroll
            for (int nh = 0; nh < 2; ++nh) {
                v8f s;
#pragma unroll
                for (int r = 0; r < 8; ++r) s[r] = 0.0f;
#pragma unroll
                for (int j = 0; j < 3; ++j) {
                    Frag16 kf;
#pragma unroll
                    for (int i = 0; i < 8; ++i)
                        kf.u[i] = Ksh32[(nh * 16 + lm) * 48 + kidx(i, half) + 16 * j];
                    s = __builtin_amdgcn_wmma_f32_16x16x32_bf16(
                        false, qf[j].v, false, kf.v, (short)0, s, false, false);
                }
                int key = kb * 32 + nh * 16 + lm;
#pragma unroll
                for (int r = 0; r < 8; ++r) {
                    int qr = q0 + r + 8 * half;
                    float v = s[r] * sc;
                    st[nh][r] = (key <= qr) ? v : -__builtin_inff();
                }
            }
            // ---- online softmax (row reductions across 16 lanes) ----
            float fr[8];
#pragma unroll
            for (int r = 0; r < 8; ++r) {
                float mx = fmaxf(st[0][r], st[1][r]);
#pragma unroll
                for (int off = 1; off < 16; off <<= 1)
                    mx = fmaxf(mx, __shfl_xor(mx, off, 16));
                float mnew = fmaxf(m_s[r], mx);
                fr[r] = __expf(m_s[r] - mnew);
                m_s[r] = mnew;
                float p0 = __expf(st[0][r] - mnew);
                float p1 = __expf(st[1][r] - mnew);
                st[0][r] = p0; st[1][r] = p1;
                float rs = p0 + p1;
#pragma unroll
                for (int off = 1; off < 16; off <<= 1)
                    rs += __shfl_xor(rs, off, 16);
                l_s[r] = l_s[r] * fr[r] + rs;
            }
#pragma unroll
            for (int nt = 0; nt < 6; ++nt)
#pragma unroll
                for (int r = 0; r < 8; ++r) acc[nt][r] *= fr[r];

            // ---- redistribute P (D-layout -> A-layout) via per-wave LDS ----
#pragma unroll
            for (int nh = 0; nh < 2; ++nh)
#pragma unroll
                for (int r = 0; r < 8; ++r)
                    Pw[(r + 8 * half) * 32 + nh * 16 + lm] = f2bf(st[nh][r]);
            asm volatile("s_wait_dscnt 0" ::: "memory");

            Frag16 pf;
#pragma unroll
            for (int i = 0; i < 8; ++i)
                pf.u[i] = ((const unsigned int*)Pw)[lm * 16 + kidx(i, half)];

            // ---- acc += P @ V  (6 n-tiles over hs=96) ----
#pragma unroll
            for (int nt = 0; nt < 6; ++nt) {
                Frag16 vf;
#pragma unroll
                for (int i = 0; i < 8; ++i)
                    vf.u[i] = Vtsh32[(nt * 16 + lm) * 16 + kidx(i, half)];
                acc[nt] = __builtin_amdgcn_wmma_f32_16x16x32_bf16(
                    false, pf.v, false, vf.v, (short)0, acc[nt], false, false);
            }
        }
        __syncthreads();
    }

    // ---- epilogue: y = acc / (l + 1e-9), bf16 into [B*T, C] ----
#pragma unroll
    for (int r = 0; r < 8; ++r) {
        float inv = 1.0f / (l_s[r] + 1e-9f);
        int t  = q0 + r + 8 * half;
        size_t rowbase = ((size_t)b * TT + t) * N_EMBD + h * HS;
#pragma unroll
        for (int nt = 0; nt < 6; ++nt)
            Y[rowbase + nt * 16 + lm] = f2bf(acc[nt][r] * inv);
    }
}

// ---------------------------------------------------------------------------
// Launch
// ---------------------------------------------------------------------------
extern "C" void kernel_launch(void* const* d_in, const int* in_sizes, int n_in,
                              void* d_out, int out_size, void* d_ws, size_t ws_size,
                              hipStream_t stream) {
    const float* x      = (const float*)d_in[0];
    const float* W_attn = (const float*)d_in[1];
    const float* b_attn = (const float*)d_in[2];
    const float* W_proj = (const float*)d_in[3];
    const float* b_proj = (const float*)d_in[4];
    float* out = (float*)d_out;

    char* ws = (char*)d_ws;
    size_t off = 0;
    unsigned short* xbf  = (unsigned short*)(ws + off); off += (size_t)MTOT * N_EMBD * 2;
    unsigned short* wabf = (unsigned short*)(ws + off); off += (size_t)N_EMBD * 3 * N_EMBD * 2;
    unsigned short* wpbf = (unsigned short*)(ws + off); off += (size_t)N_EMBD * N_EMBD * 2;
    unsigned short* Qb   = (unsigned short*)(ws + off); off += (size_t)MTOT * N_EMBD * 2;
    unsigned short* Kbuf = (unsigned short*)(ws + off); off += (size_t)MTOT * N_EMBD * 2;
    unsigned short* Vbuf = (unsigned short*)(ws + off); off += (size_t)MTOT * N_EMBD * 2;
    unsigned short* Yb   = (unsigned short*)(ws + off); off += (size_t)MTOT * N_EMBD * 2;

    // 1) fp32 -> bf16 conversions; weights transposed to [N][K]
    cvt_bf16_kernel<<<4096, 256, 0, stream>>>(x, xbf, MTOT * N_EMBD);
    cvt_transpose_kernel<<<dim3((3 * N_EMBD) / 32, N_EMBD / 32), dim3(32, 8), 0, stream>>>(
        W_attn, wabf, N_EMBD, 3 * N_EMBD);
    cvt_transpose_kernel<<<dim3(N_EMBD / 32, N_EMBD / 32), dim3(32, 8), 0, stream>>>(
        W_proj, wpbf, N_EMBD, N_EMBD);

    // 2) qkv projection: [8192,768] x [768,2304] -> q/k [B,H,T,hs], v [B,H,hs,T]
    gemm_bf16_kernel<<<dim3(MTOT / 128, (3 * N_EMBD) / 128), 256, 0, stream>>>(
        xbf, wabf, b_attn, 3 * N_EMBD, N_EMBD, 0, Qb, Kbuf, Vbuf, nullptr);

    // 3) causal flash attention -> Y bf16 [8192,768]
    attn_kernel<<<dim3(BB * N_HEAD, TT / 128), 256, 0, stream>>>(Qb, Kbuf, Vbuf, Yb);

    // 4) output projection: [8192,768] x [768,768] + bias -> fp32 out
    gemm_bf16_kernel<<<dim3(MTOT / 128, N_EMBD / 128), 256, 0, stream>>>(
        Yb, wpbf, b_proj, N_EMBD, N_EMBD, 1, nullptr, nullptr, nullptr, out);
}
